// ShiTomasiBADSinkhornMatcher_23562190586070
// MI455X (gfx1250) — matrix operand order repaired and checked by reading, not compiled, not run on hardware
//
#include <hip/hip_runtime.h>
#include <hip/hip_bf16.h>
#include <stdint.h>

#define IMH 480
#define IMW 640
#define NPIX (IMH*IMW)
#define KPTS 512
#define DPAIRS 256
#define CAP 32768
#define N1 4096
#define S1BLOCKS (CAP / N1)       // 8
#define N2 (S1BLOCKS * KPTS)      // 4096
#define NSINK 20
#define MSZ 513
#define NORMC (-6.93147180559945309f)   // -log(2K) = -log(1024)
#define LOGMU_BIN (-0.69314718055994531f) // log(K) - log(2K) = -log 2

typedef _Float16 half_t;
typedef _Float16 v16h __attribute__((ext_vector_type(16)));
typedef _Float16 v8h  __attribute__((ext_vector_type(8)));
typedef float    v8f  __attribute__((ext_vector_type(8)));
typedef unsigned long long u64;

// ---------------- utility ----------------
__device__ __forceinline__ float img_at(const float* im, int y, int x) {
    if (y < 0 || y >= IMH || x < 0 || x >= IMW) return 0.f;
    return im[y * IMW + x];
}
__device__ __forceinline__ int clampi(int v, int lo, int hi) {
    return v < lo ? lo : (v > hi ? hi : v);
}

// ---------------- MT19937 (numpy RandomState(42)) -> BAD offsets ----------------
__device__ unsigned int mt_next(unsigned int* mt, int* mti) {
    if (*mti >= 624) {
        for (int k = 0; k < 624; ++k) {
            unsigned int y = (mt[k] & 0x80000000u) | (mt[(k + 1) % 624] & 0x7fffffffu);
            unsigned int v = mt[(k + 397) % 624] ^ (y >> 1);
            if (y & 1u) v ^= 2567483615u;
            mt[k] = v;
        }
        *mti = 0;
    }
    unsigned int y = mt[(*mti)++];
    y ^= y >> 11;
    y ^= (y << 7)  & 2636928640u;
    y ^= (y << 15) & 4022730752u;
    y ^= y >> 18;
    return y;
}

__global__ void mt_offsets_kernel(int* offs) {
    __shared__ unsigned int mt[624];
    if (threadIdx.x != 0 || blockIdx.x != 0) return;
    mt[0] = 42u;
    for (int i = 1; i < 624; ++i)
        mt[i] = 1812433253u * (mt[i - 1] ^ (mt[i - 1] >> 30)) + (unsigned int)i;
    int mti = 624;
    for (int p = 0; p < DPAIRS * 4; ++p) {
        unsigned int a = mt_next(mt, &mti) >> 5;
        unsigned int b = mt_next(mt, &mti) >> 6;
        double u = ((double)a * 67108864.0 + (double)b) / 9007199254740992.0;
        double val = -8.0 + 16.0 * u;           // uniform(-PS/2, PS/2)
        offs[p] = (int)rint(val);               // round half-to-even (np.round)
    }
}

// ---------------- detector ----------------
__global__ void grad_kernel(const float* __restrict__ img, float* Ix, float* Iy) {
    int i = blockIdx.x * blockDim.x + threadIdx.x;
    if (i >= NPIX) return;
    int y = i / IMW, x = i % IMW;
    float a = img_at(img, y-1, x-1), b = img_at(img, y-1, x), c = img_at(img, y-1, x+1);
    float d = img_at(img, y,   x-1),                          f = img_at(img, y,   x+1);
    float g = img_at(img, y+1, x-1), h = img_at(img, y+1, x), k = img_at(img, y+1, x+1);
    // cross-correlation with kx=[[-1,0,1],[-2,0,2],[-1,0,1]], ky = kx^T
    Ix[i] = -a + c - 2.f*d + 2.f*f - g + k;
    Iy[i] = -a - 2.f*b - c + g + 2.f*h + k;
}

__global__ void score_kernel(const float* __restrict__ Ix, const float* __restrict__ Iy,
                             float* score) {
    int i = blockIdx.x * blockDim.x + threadIdx.x;
    if (i >= NPIX) return;
    int y = i / IMW, x = i % IMW;
    float sxx = 0.f, syy = 0.f, sxy = 0.f;
    for (int dy = -1; dy <= 1; ++dy)
        for (int dx = -1; dx <= 1; ++dx) {
            int yy = y + dy, xx = x + dx;
            if (yy < 0 || yy >= IMH || xx < 0 || xx >= IMW) continue;
            float ix = Ix[yy * IMW + xx], iy = Iy[yy * IMW + xx];
            sxx += ix * ix; syy += iy * iy; sxy += ix * iy;
        }
    sxx *= (1.f / 9.f); syy *= (1.f / 9.f); sxy *= (1.f / 9.f);
    float half_tr = 0.5f * (sxx + syy);
    float hd = 0.5f * (sxx - syy);
    score[i] = half_tr - sqrtf(hd * hd + sxy * sxy + 1e-12f);
}

__global__ void smooth_kernel(const float* __restrict__ img, float* smooth) {
    int i = blockIdx.x * blockDim.x + threadIdx.x;
    if (i >= NPIX) return;
    int y = i / IMW, x = i % IMW;
    float s = 0.f;
    for (int dy = -2; dy <= 2; ++dy)
        for (int dx = -2; dx <= 2; ++dx)
            s += img_at(img, y + dy, x + dx);
    smooth[i] = s * (1.f / 25.f);
}

// ---------------- NMS + compaction ----------------
__global__ void zero_cand_kernel(u64* cand, int* counter) {
    int i = blockIdx.x * blockDim.x + threadIdx.x;
    if (i < CAP) cand[i] = 0ull;
    if (i == 0) *counter = 0;
}

__global__ void nms_compact_kernel(const float* __restrict__ score, u64* cand, int* counter) {
    int i = blockIdx.x * blockDim.x + threadIdx.x;
    if (i >= NPIX) return;
    float s = score[i];
    if (s <= 0.f) return;                        // THRESH = 0
    int y = i / IMW, x = i % IMW;
    float m = -1e30f;
    for (int dy = -3; dy <= 3; ++dy) {
        int yy = y + dy;
        if (yy < 0 || yy >= IMH) continue;
        for (int dx = -3; dx <= 3; ++dx) {
            int xx = x + dx;
            if (xx < 0 || xx >= IMW) continue;
            m = fmaxf(m, score[yy * IMW + xx]);
        }
    }
    if (s >= m - 1e-7f) {
        int slot = atomicAdd(counter, 1);
        if (slot < CAP) {
            unsigned int sb = __float_as_uint(s);          // s>0 -> monotonic bits
            cand[slot] = ((u64)sb << 32) | (unsigned int)(~(unsigned int)i);
        }
    }
}

// ---------------- bitonic top-K ----------------
__device__ void bitonic_desc(u64* keys, int n, int tid, int nthr) {
    for (int k = 2; k <= n; k <<= 1) {
        for (int j = k >> 1; j > 0; j >>= 1) {
            for (int i = tid; i < n; i += nthr) {
                int ixj = i ^ j;
                if (ixj > i) {
                    u64 a = keys[i], b = keys[ixj];
                    bool flip = ((i & k) == 0);            // descending overall
                    if (flip == (a < b)) { keys[i] = b; keys[ixj] = a; }
                }
            }
            __syncthreads();
        }
    }
}

__global__ void sort_stage1_kernel(const u64* __restrict__ cand, u64* stage) {
    __shared__ u64 keys[N1];
    int base = blockIdx.x * N1;
    for (int i = threadIdx.x; i < N1; i += blockDim.x) keys[i] = cand[base + i];
    __syncthreads();
    bitonic_desc(keys, N1, threadIdx.x, blockDim.x);
    for (int i = threadIdx.x; i < KPTS; i += blockDim.x)
        stage[blockIdx.x * KPTS + i] = keys[i];
}

__global__ void sort_stage2_kernel(const u64* __restrict__ stage, float* kp_out, int* kpint) {
    __shared__ u64 keys[N2];
    for (int i = threadIdx.x; i < N2; i += blockDim.x) keys[i] = stage[i];
    __syncthreads();
    bitonic_desc(keys, N2, threadIdx.x, blockDim.x);
    int t = threadIdx.x;
    if (t < KPTS) {
        u64 key = keys[t];
        if ((unsigned int)(key >> 32) != 0u) {
            unsigned int idx = ~(unsigned int)key;
            int y = (int)(idx / IMW), x = (int)(idx % IMW);
            kp_out[2 * t] = (float)y; kp_out[2 * t + 1] = (float)x;
            kpint[2 * t] = y;         kpint[2 * t + 1] = x;
        } else {
            kp_out[2 * t] = -1.f; kp_out[2 * t + 1] = -1.f;
            kpint[2 * t] = -1;    kpint[2 * t + 1] = -1;
        }
    }
}

// ---------------- descriptors (BAD at keypoints, L2-normalized, fp16 copy) -------
__global__ void desc_kernel(const float* __restrict__ smooth, const int* __restrict__ kpint,
                            const int* __restrict__ offs, half_t* dh, float* a2) {
    int k = blockIdx.x;        // keypoint
    int t = threadIdx.x;       // pair (256)
    int y = kpint[2 * k], x = kpint[2 * k + 1];
    float val = 0.f;
    if (y >= 0) {   // edge-replicate pad == coordinate clamp
        int o0 = offs[4*t], o1 = offs[4*t+1], o2 = offs[4*t+2], o3 = offs[4*t+3];
        int y1 = clampi(y + o0, 0, IMH - 1), x1 = clampi(x + o1, 0, IMW - 1);
        int y2 = clampi(y + o2, 0, IMH - 1), x2 = clampi(x + o3, 0, IMW - 1);
        val = smooth[y1 * IMW + x1] - smooth[y2 * IMW + x2];
    }
    __shared__ float red[DPAIRS];
    red[t] = val * val;
    __syncthreads();
    for (int s = DPAIRS / 2; s > 0; s >>= 1) {
        if (t < s) red[t] += red[t + s];
        __syncthreads();
    }
    float ss = red[0];
    float inv = 1.f / (sqrtf(ss) + 1e-8f);
    dh[(size_t)k * DPAIRS + t] = (half_t)(val * inv);
    if (t == 0) a2[k] = ss * inv * inv;      // ||d||^2 after normalization
}

// ---------------- WMMA GEMM: sim = -sqrt(max(a2+b2-2*d1.d2^T,0)+1e-12) ----------
// One 16x16 output tile per wave32; A/B in contiguous k order (k = 16h + s), C per CDNA5 ISA 7.12.2.
__global__ void gemm_sim_kernel(const half_t* __restrict__ d1h, const half_t* __restrict__ d2h,
                                const float* __restrict__ a2, const float* __restrict__ b2,
                                float* __restrict__ Z) {
    int lane = threadIdx.x & 31;
    int wave = threadIdx.x >> 5;
    int tile = blockIdx.x * 8 + wave;          // 32x32 = 1024 tiles
    int tm = (tile >> 5) << 4;
    int tn = (tile & 31) << 4;
    int l  = lane & 15;
    bool hi = lane >= 16;

    const half_t* pA = d1h + (size_t)(tm + l) * DPAIRS;  // A row (M)
    const half_t* pB = d2h + (size_t)(tn + l) * DPAIRS;  // B col (N) = d2 row

    v8f c = {};
#pragma unroll
    for (int kk = 0; kk < DPAIRS; kk += 32) {
        // A 16x32: lanes<16 hold K kk..kk+15 of row M=l; lanes>=16 K kk+16..kk+31
        v16h a = *(const v16h*)(pA + kk + (hi ? 16 : 0));
        // B 32x16: lanes<16 hold K kk..kk+15 of col N=l; lanes>=16 K kk+16..kk+31
        v16h b = *(const v16h*)(pB + kk + (hi ? 16 : 0));
        if (kk + 32 < DPAIRS) __builtin_prefetch(pA + kk + 32, 0, 0);
        c = __builtin_amdgcn_wmma_f32_16x16x32_f16(false, a, false, b,
                                                   (short)0, c, false, false);
    }
    int n = tn + l;
    float bn = b2[n];
#pragma unroll
    for (int r = 0; r < 8; ++r) {
        int m = tm + r + (hi ? 8 : 0);
        float dot = c[r];
        float sq = fmaxf(a2[m] + bn - 2.f * dot, 0.f);
        Z[(size_t)m * MSZ + n] = -sqrtf(sq + 1e-12f);    // EPSILON == 1
    }
}

// ---------------- Sinkhorn ----------------
__global__ void sinkhorn_init_kernel(float* Z, float* u, float* v) {
    int j = blockIdx.x * blockDim.x + threadIdx.x;
    if (j < MSZ) {
        Z[(size_t)KPTS * MSZ + j] = 1.0f;   // dustbin row
        Z[(size_t)j * MSZ + KPTS] = 1.0f;   // dustbin col
        u[j] = 0.f; v[j] = 0.f;
    }
}

// axis==0: u[i] = log_mu[i] - LSE_j(Z[i,j]+v[j]);  axis==1: columns, other=u
__global__ void lse_update_kernel(const float* __restrict__ Z, const float* __restrict__ other,
                                  float* target, int axis) {
    int i = blockIdx.x;          // 0..512
    int t = threadIdx.x;         // 256
    const float* base = (axis == 0) ? (Z + (size_t)i * MSZ) : (Z + i);
    size_t step = (axis == 0) ? 1 : MSZ;
    float m = -1e30f;
    for (int j = t; j < MSZ; j += 256) m = fmaxf(m, base[(size_t)j * step] + other[j]);
    __shared__ float red[256];
    red[t] = m; __syncthreads();
    for (int s = 128; s > 0; s >>= 1) { if (t < s) red[t] = fmaxf(red[t], red[t + s]); __syncthreads(); }
    m = red[0]; __syncthreads();
    float acc = 0.f;
    for (int j = t; j < MSZ; j += 256) acc += expf(base[(size_t)j * step] + other[j] - m);
    red[t] = acc; __syncthreads();
    for (int s = 128; s > 0; s >>= 1) { if (t < s) red[t] += red[t + s]; __syncthreads(); }
    if (t == 0) {
        float logmu = (i < KPTS) ? NORMC : LOGMU_BIN;
        target[i] = logmu - (m + logf(red[0]));
    }
}

__global__ void probs_kernel(const float* __restrict__ Z, const float* __restrict__ u,
                             const float* __restrict__ v, float* out) {
    int idx = blockIdx.x * blockDim.x + threadIdx.x;
    if (idx >= MSZ * MSZ) return;
    int i = idx / MSZ, j = idx % MSZ;
    out[idx] = expf(Z[idx] + u[i] + v[j] - NORMC);
}

// ---------------- orchestration ----------------
extern "C" void kernel_launch(void* const* d_in, const int* in_sizes, int n_in,
                              void* d_out, int out_size, void* d_ws, size_t ws_size,
                              hipStream_t stream) {
    const float* imgs[2] = { (const float*)d_in[0], (const float*)d_in[1] };
    float* out = (float*)d_out;

    char* ws = (char*)d_ws;
    size_t off = 0;
    auto alloc = [&](size_t bytes) -> void* {
        off = (off + 255) & ~(size_t)255;
        void* p = ws + off;
        off += bytes;
        return p;
    };
    int*    offs    = (int*)   alloc(DPAIRS * 4 * sizeof(int));
    float*  Ix      = (float*) alloc(NPIX * sizeof(float));
    float*  Iy      = (float*) alloc(NPIX * sizeof(float));
    float*  score   = (float*) alloc(NPIX * sizeof(float));
    float*  smooth  = (float*) alloc(NPIX * sizeof(float));
    u64*    cand    = (u64*)   alloc(CAP * sizeof(u64));
    u64*    stage   = (u64*)   alloc(N2 * sizeof(u64));
    int*    counter = (int*)   alloc(64);
    int*    kpint   = (int*)   alloc(2 * KPTS * 2 * sizeof(int));
    half_t* d1h     = (half_t*)alloc((size_t)KPTS * DPAIRS * sizeof(half_t));
    half_t* d2h     = (half_t*)alloc((size_t)KPTS * DPAIRS * sizeof(half_t));
    float*  a2      = (float*) alloc(KPTS * sizeof(float));
    float*  b2      = (float*) alloc(KPTS * sizeof(float));
    float*  Zm      = (float*) alloc((size_t)MSZ * MSZ * sizeof(float));
    float*  uvec    = (float*) alloc(MSZ * sizeof(float));
    float*  vvec    = (float*) alloc(MSZ * sizeof(float));
    (void)ws_size; (void)in_sizes; (void)n_in; (void)out_size;

    const int TPB = 256;
    const int pixBlocks = (NPIX + TPB - 1) / TPB;

    mt_offsets_kernel<<<1, 32, 0, stream>>>(offs);

    for (int im = 0; im < 2; ++im) {
        const float* img = imgs[im];
        half_t* dh  = im ? d2h : d1h;
        float*  aa  = im ? b2  : a2;
        int*    kpi = kpint + im * KPTS * 2;
        float*  kpo = out + im * KPTS * 2;

        grad_kernel  <<<pixBlocks, TPB, 0, stream>>>(img, Ix, Iy);
        score_kernel <<<pixBlocks, TPB, 0, stream>>>(Ix, Iy, score);
        smooth_kernel<<<pixBlocks, TPB, 0, stream>>>(img, smooth);
        zero_cand_kernel<<<(CAP + TPB - 1) / TPB, TPB, 0, stream>>>(cand, counter);
        nms_compact_kernel<<<pixBlocks, TPB, 0, stream>>>(score, cand, counter);
        sort_stage1_kernel<<<S1BLOCKS, 1024, 0, stream>>>(cand, stage);
        sort_stage2_kernel<<<1, 1024, 0, stream>>>(stage, kpo, kpi);
        desc_kernel<<<KPTS, DPAIRS, 0, stream>>>(smooth, kpi, offs, dh, aa);
    }

    gemm_sim_kernel<<<(32 * 32) / 8, 256, 0, stream>>>(d1h, d2h, a2, b2, Zm);
    sinkhorn_init_kernel<<<(MSZ + TPB - 1) / TPB, TPB, 0, stream>>>(Zm, uvec, vvec);
    for (int it = 0; it < NSINK; ++it) {
        lse_update_kernel<<<MSZ, 256, 0, stream>>>(Zm, vvec, uvec, 0);
        lse_update_kernel<<<MSZ, 256, 0, stream>>>(Zm, uvec, vvec, 1);
    }
    probs_kernel<<<(MSZ * MSZ + TPB - 1) / TPB, TPB, 0, stream>>>(
        Zm, uvec, vvec, out + 2 * KPTS * 2);
}
